// ALiBiMultiHeadAttention_67645734912302
// MI455X (gfx1250) — compile-verified, matrix operand order checked by reading, not run on hardware
//
#include <hip/hip_runtime.h>
#include <hip/hip_bf16.h>
#include <math.h>
#include <stdint.h>

// ---------------------------------------------------------------------------
// ALiBi multi-head attention, MI455X (gfx1250, wave32)
// WMMA 16x16x32 f16 + Tensor Data Mover (TDM) tile staging.
// B=2, S=2048, D=1024, H=16, hd=64
// ---------------------------------------------------------------------------

#define BATCH 2
#define SEQ   2048
#define DMODEL 1024
#define NHEADS 16
#define HDIM  64
#define MROWS (BATCH * SEQ)   // 4096

typedef _Float16 v4h  __attribute__((ext_vector_type(4)));
typedef _Float16 v8h  __attribute__((ext_vector_type(8)));
typedef _Float16 v16h __attribute__((ext_vector_type(16)));
typedef float    v8f  __attribute__((ext_vector_type(8)));

#define SHUF16(lo, hi) __builtin_shufflevector((lo), (hi), 0,1,2,3,4,5,6,7,8,9,10,11,12,13,14,15)

// ---------------------------------------------------------------------------
// Tensor Data Mover: one wave DMAs a 2D f16 tile (tile_w x tile_h elements,
// global row stride row_stride_elems) into LDS. Tracked with TENSORcnt.
// D# layout per CDNA5 ISA (group0/group1; groups 2,3 zero
// for <=2D tensors). tensor_dim* set equal to the tile so no OOB clipping.
// ---------------------------------------------------------------------------
#if __has_builtin(__builtin_amdgcn_tensor_load_to_lds)
#define HAVE_TDM 1
typedef unsigned int u32x4 __attribute__((ext_vector_type(4)));
typedef int          i32x4 __attribute__((ext_vector_type(4)));
typedef int          i32x8 __attribute__((ext_vector_type(8)));

__device__ __forceinline__ void tdm_load_tile_f16(const _Float16* gsrc,
                                                  void* lds_dst,
                                                  unsigned tile_w,
                                                  unsigned tile_h,
                                                  unsigned row_stride_elems) {
    unsigned long long ga = (unsigned long long)(uintptr_t)gsrc;
    unsigned lds_addr = (unsigned)(uintptr_t)lds_dst;   // LDS aperture: addr[31:0]
    // group0: [1:0]=count=1, [63:32]=lds_addr, [120:64]=global_addr, [127:126]=type 2
    u32x4 g0 = { 1u,
                 lds_addr,
                 (unsigned)(ga & 0xFFFFFFFFu),
                 ((unsigned)((ga >> 32) & 0x01FFFFFFu)) | 0x80000000u };
    // group1: [17:16]=data_size(2B), [79:48]=tensor_dim0, [111:80]=tensor_dim1,
    //         [127:112]=tile_dim0, [143:128]=tile_dim1, [207:160]=tensor_dim0_stride
    i32x8 g1 = { (int)(1u << 16),
                 (int)((tile_w & 0xFFFFu) << 16),
                 (int)((tile_h & 0xFFFFu) << 16),
                 (int)((tile_w & 0xFFFFu) << 16),
                 (int)(tile_h & 0xFFFFu),
                 (int)row_stride_elems,
                 0, 0 };
    i32x4 gz = { 0, 0, 0, 0 };
#if defined(__clang_major__) && (__clang_major__ >= 23)
    i32x8 gz8 = { 0, 0, 0, 0, 0, 0, 0, 0 };
    __builtin_amdgcn_tensor_load_to_lds(g0, g1, gz, gz, gz8, 0);
#else
    __builtin_amdgcn_tensor_load_to_lds(g0, g1, gz, gz, 0);
#endif
}

__device__ __forceinline__ void tdm_wait() {
#if __has_builtin(__builtin_amdgcn_s_wait_tensorcnt)
    __builtin_amdgcn_s_wait_tensorcnt(0);
#endif
}
#else
#define HAVE_TDM 0
#endif

// --------------------------------------------------------------------------
// f32 -> f16 cast (vectorized 4-wide); n is always a multiple of 4 here.
// --------------------------------------------------------------------------
__global__ __launch_bounds__(256) void cast_f32_to_f16(const float* __restrict__ src,
                                                       _Float16* __restrict__ dst, int n) {
    int i = (blockIdx.x * blockDim.x + threadIdx.x) * 4;
    if (i < n) {
        float4 f = *(const float4*)(src + i);
        v4h o;
        o.x = (_Float16)f.x; o.y = (_Float16)f.y;
        o.z = (_Float16)f.z; o.w = (_Float16)f.w;
        *(v4h*)(dst + i) = o;
    }
}

// --------------------------------------------------------------------------
// WMMA GEMM:  C[M=4096, N=1024] = X[4096,1024] * W[1024,1024]  (f16 in, f32 acc)
// mode 0: write f16 head-split [B, H, S, hd];  mode 1: write f32 [M, N]
// Block: 256 threads = 8 waves; tile 64(M) x 128(N); BK = 32.
// A tile staged by the TDM (one descriptor per K-step, issued by wave 0);
// B tile transposed into LDS by all threads (TDM cannot transpose).
// --------------------------------------------------------------------------
__global__ __launch_bounds__(256) void gemm_wmma_f16(const _Float16* __restrict__ X,
                                                     const _Float16* __restrict__ W,
                                                     _Float16* __restrict__ outHS,
                                                     float* __restrict__ outF,
                                                     int mode) {
    __shared__ __align__(16) _Float16 Atile[64][32];   // 64 rows x 32 K (row = 64B)
    __shared__ __align__(16) _Float16 Bt[128][40];     // transposed B: [n][k] (+pad)

    const int t    = threadIdx.x;
    const int wid  = t >> 5;
    const int lane = t & 31;
    const int half = lane >> 4;
    const int l16  = lane & 15;

    const int m0 = blockIdx.x * 64;
    const int n0 = blockIdx.y * 128;

    v8f acc[4] = {};

    for (int k0 = 0; k0 < DMODEL; k0 += 32) {
        __syncthreads();
        // ---- stage A tile via Tensor Data Mover (64 x 32 f16, stride 1024) ----
#if HAVE_TDM
        if (wid == 0) {
            tdm_load_tile_f16(&X[(size_t)m0 * DMODEL + k0], &Atile[0][0],
                              /*tile_w=*/32, /*tile_h=*/64, /*stride=*/DMODEL);
        }
#else
        {
            int r = t >> 2, c = t & 3;
            *(v8h*)&Atile[r][c * 8] =
                *(const v8h*)&X[(size_t)(m0 + r) * DMODEL + k0 + c * 8];
        }
#endif
        // ---- stage B tile transposed: Bt[n][k] = W[k0+k][n0+n] ----
        {
            int k = t & 31, nc = t >> 5;  // 8 waves -> 8 chunks of 16 columns
            const _Float16* s = &W[(size_t)(k0 + k) * DMODEL + n0 + nc * 16];
            v8h a = *(const v8h*)s;
            v8h b = *(const v8h*)(s + 8);
#pragma unroll
            for (int i = 0; i < 8; i++) {
                Bt[nc * 16 + i][k]     = a[i];
                Bt[nc * 16 + 8 + i][k] = b[i];
            }
        }
        if (k0 + 32 < DMODEL)   // gfx1250 global_prefetch_b8 for next W tile
            __builtin_prefetch(&W[(size_t)(k0 + 32 + (t & 31)) * DMODEL + n0], 0, 0);
#if HAVE_TDM
        if (wid == 0) tdm_wait();   // TENSORcnt -> 0 before the barrier opens LDS
#endif
        __syncthreads();

        // ---- A fragment: row = (wid&3)*16 + l16; ISA lane-half K pattern ----
        int arow = (wid & 3) * 16 + l16;
        v8h alo = *(v8h*)&Atile[arow][half * 8];
        v8h ahi = *(v8h*)&Atile[arow][16 + half * 8];
        v16h af = SHUF16(alo, ahi);

#pragma unroll
        for (int c = 0; c < 4; c++) {
            int bcol = (wid >> 2) * 64 + c * 16 + l16;
            v8h blo = *(v8h*)&Bt[bcol][half * 8];
            v8h bhi = *(v8h*)&Bt[bcol][16 + half * 8];
            v16h bf = SHUF16(blo, bhi);
            acc[c] = __builtin_amdgcn_wmma_f32_16x16x32_f16(
                false, af, false, bf, (short)0, acc[c], false, false);
        }
    }

    // ---- epilogue: C layout is VGPR r -> row r (+8 for upper half), col = l16 ----
#pragma unroll
    for (int c = 0; c < 4; c++) {
#pragma unroll
        for (int r = 0; r < 8; r++) {
            int m = m0 + (wid & 3) * 16 + r + 8 * half;
            int n = n0 + (wid >> 2) * 64 + c * 16 + l16;
            float v = acc[c][r];
            if (mode == 0) {
                int bb = m >> 11, srow = m & (SEQ - 1);
                int hh = n >> 6,  dd   = n & (HDIM - 1);
                outHS[(((size_t)bb * NHEADS + hh) * SEQ + srow) * HDIM + dd] = (_Float16)v;
            } else {
                outF[(size_t)m * DMODEL + n] = v;
            }
        }
    }
}

// --------------------------------------------------------------------------
// Flash attention with ALiBi. Q,K,V in [B,H,S,hd] f16.
// grid = (B*H, S/64); block = 128 (4 waves); each wave owns 16 query rows.
// KV tiles of 32 keys; K tile staged by TDM; V tile transposed manually.
// QK^T = 4 WMMAs (2 tiles x 2 K-steps); PV = 4 WMMAs (hd = 64 = 4x16) per iteration.
// --------------------------------------------------------------------------
__global__ __launch_bounds__(128) void attn_alibi_wmma(const _Float16* __restrict__ Qh,
                                                       const _Float16* __restrict__ Kh,
                                                       const _Float16* __restrict__ Vh,
                                                       _Float16* __restrict__ ctx) {
    __shared__ __align__(16) _Float16 Kt[32][64];        // 32 keys x 64 dims (row = 128B)
    __shared__ __align__(16) _Float16 Vt[64][40];        // transposed: [d][k] (+pad)
    __shared__ __align__(16) _Float16 Ptile[4][16][40];  // per-wave P staging

    const int t    = threadIdx.x;
    const int wid  = t >> 5;
    const int lane = t & 31;
    const int half = lane >> 4;
    const int l16  = lane & 15;

    const int bh = blockIdx.x;           // b*H + h
    const int h  = bh & (NHEADS - 1);
    const int b  = bh >> 4;
    const int q0 = blockIdx.y * 64;

    const float slope = exp2f(-0.5f * (float)(h + 1));
    const size_t headBase = (size_t)bh * SEQ;            // row offset in [B*H, S, hd]

    // ---- load Q fragments once (row-major, contiguous hd=64) ----
    size_t qrow = headBase + (size_t)(q0 + wid * 16 + l16);
    const _Float16* qp = &Qh[qrow * HDIM];
    v16h qf[2];
#pragma unroll
    for (int j = 0; j < 2; j++) {
        v8h lo = *(const v8h*)&qp[j * 32 + half * 8];
        v8h hi = *(const v8h*)&qp[j * 32 + 16 + half * 8];
        qf[j] = SHUF16(lo, hi);
    }

    float m_i[8], l_i[8];
#pragma unroll
    for (int r = 0; r < 8; r++) { m_i[r] = -3.0e38f; l_i[r] = 0.0f; }
    v8f accO[4] = {};

    for (int kv0 = 0; kv0 < SEQ; kv0 += 32) {
        __syncthreads();
        // ---- stage K tile via TDM (32 x 64 f16, stride 64) ----
#if HAVE_TDM
        if (wid == 0) {
            tdm_load_tile_f16(&Kh[(headBase + kv0) * HDIM], &Kt[0][0],
                              /*tile_w=*/HDIM, /*tile_h=*/32, /*stride=*/HDIM);
        }
#else
        if (wid < 1) { /* unreachable placeholder to keep structure */ }
        {
            int row = t >> 2, c = t & 3;
            const _Float16* s = &Kh[(headBase + kv0 + row) * HDIM + c * 16];
            *(v8h*)&Kt[row][c * 16]     = *(const v8h*)s;
            *(v8h*)&Kt[row][c * 16 + 8] = *(const v8h*)(s + 8);
        }
#endif
        // ---- stage V tile transposed: Vt[d][k] ----
        {
            int kk = t & 31, dc = t >> 5;
            const _Float16* s = &Vh[(headBase + kv0 + kk) * HDIM + dc * 16];
            v8h a = *(const v8h*)s;
            v8h c2 = *(const v8h*)(s + 8);
#pragma unroll
            for (int i = 0; i < 8; i++) {
                Vt[dc * 16 + i][kk]     = a[i];
                Vt[dc * 16 + 8 + i][kk] = c2[i];
            }
        }
#if HAVE_TDM
        if (wid == 0) tdm_wait();
#endif
        __syncthreads();

        // ---- scores: two 16x16 tiles over this 32-key block ----
        v8f sc[2] = {};
#pragma unroll
        for (int t2 = 0; t2 < 2; t2++) {
            int krow = t2 * 16 + l16;       // B fragment col n -> key row (contiguous in Kt)
#pragma unroll
            for (int j = 0; j < 2; j++) {   // hd = 64 -> two K-steps of 32
                v8h blo = *(v8h*)&Kt[krow][j * 32 + half * 8];
                v8h bhi = *(v8h*)&Kt[krow][j * 32 + 16 + half * 8];
                v16h bf = SHUF16(blo, bhi);
                sc[t2] = __builtin_amdgcn_wmma_f32_16x16x32_f16(
                    false, qf[j], false, bf, (short)0, sc[t2], false, false);
            }
        }

        // ---- ALiBi bias + online softmax (rows replicated across 16-lane halves) ----
        const float j0 = (float)(kv0 + l16);
        const float j1 = (float)(kv0 + 16 + l16);
        float p0[8], p1[8];
#pragma unroll
        for (int r = 0; r < 8; r++) {
            float irow = (float)(q0 + wid * 16 + r + 8 * half);
            float s0 = sc[0][r] * 0.125f - slope * fabsf(irow - j0);
            float s1 = sc[1][r] * 0.125f - slope * fabsf(irow - j1);
            float mx = fmaxf(s0, s1);
#pragma unroll
            for (int d = 8; d >= 1; d >>= 1) mx = fmaxf(mx, __shfl_xor(mx, d, 32));
            float mnew = fmaxf(m_i[r], mx);
            s0 = __expf(s0 - mnew);
            s1 = __expf(s1 - mnew);
            float rs = s0 + s1;
#pragma unroll
            for (int d = 8; d >= 1; d >>= 1) rs += __shfl_xor(rs, d, 32);
            float alpha = __expf(m_i[r] - mnew);
            l_i[r] = l_i[r] * alpha + rs;
            m_i[r] = mnew;
            p0[r] = s0; p1[r] = s1;
#pragma unroll
            for (int c = 0; c < 4; c++) accO[c][r] *= alpha;
        }

        // ---- transpose P (C layout: lane=col) -> A layout (lane=row) via LDS ----
#pragma unroll
        for (int r = 0; r < 8; r++) {
            Ptile[wid][r + 8 * half][l16]      = (_Float16)p0[r];
            Ptile[wid][r + 8 * half][16 + l16] = (_Float16)p1[r];
        }
        __syncthreads();   // make per-wave P writes visible for fragment reload

        v8h plo = *(v8h*)&Ptile[wid][l16][half * 8];
        v8h phi = *(v8h*)&Ptile[wid][l16][16 + half * 8];
        v16h pf = SHUF16(plo, phi);

        // ---- PV: (16x32) x (32x64) -> 4 accumulator tiles ----
#pragma unroll
        for (int c = 0; c < 4; c++) {
            int drow = c * 16 + l16;        // B col n -> output dim (contiguous in Vt)
            v8h blo = *(v8h*)&Vt[drow][half * 8];
            v8h bhi = *(v8h*)&Vt[drow][16 + half * 8];
            v16h bf = SHUF16(blo, bhi);
            accO[c] = __builtin_amdgcn_wmma_f32_16x16x32_f16(
                false, pf, false, bf, (short)0, accO[c], false, false);
        }
    }

    // ---- normalize and write context as [B, S, D] f16 ----
#pragma unroll
    for (int c = 0; c < 4; c++) {
#pragma unroll
        for (int r = 0; r < 8; r++) {
            int srow = q0 + wid * 16 + r + 8 * half;
            int dd   = c * 16 + l16;
            float o = accO[c][r] / l_i[r];
            ctx[((size_t)b * SEQ + srow) * DMODEL + h * HDIM + dd] = (_Float16)o;
        }
    }
}

// --------------------------------------------------------------------------
// Host-side orchestration
// --------------------------------------------------------------------------
extern "C" void kernel_launch(void* const* d_in, const int* in_sizes, int n_in,
                              void* d_out, int out_size, void* d_ws, size_t ws_size,
                              hipStream_t stream) {
    const float* q_f  = (const float*)d_in[0];
    const float* k_f  = (const float*)d_in[1];
    const float* v_f  = (const float*)d_in[2];
    const float* Wq_f = (const float*)d_in[3];
    const float* Wk_f = (const float*)d_in[4];
    const float* Wv_f = (const float*)d_in[5];
    const float* Wo_f = (const float*)d_in[6];
    float* out = (float*)d_out;

    const size_t nBSD = (size_t)BATCH * SEQ * DMODEL;   // 4,194,304
    const size_t nDD  = (size_t)DMODEL * DMODEL;        // 1,048,576

    _Float16* w    = (_Float16*)d_ws;
    _Float16* q16  = w;            w += nBSD;
    _Float16* k16  = w;            w += nBSD;
    _Float16* v16  = w;            w += nBSD;
    _Float16* wq16 = w;            w += nDD;
    _Float16* wk16 = w;            w += nDD;
    _Float16* wv16 = w;            w += nDD;
    _Float16* wo16 = w;            w += nDD;
    _Float16* Qh   = w;            w += nBSD;   // [B,H,S,hd]
    _Float16* Kh   = w;            w += nBSD;
    _Float16* Vh   = w;            w += nBSD;
    _Float16* ctx  = w;            w += nBSD;   // [B,S,D]

    // 1) down-convert everything to f16
    {
        dim3 blk(256);
        dim3 gBSD((unsigned)(nBSD / (256 * 4)));
        dim3 gDD((unsigned)(nDD / (256 * 4)));
        cast_f32_to_f16<<<gBSD, blk, 0, stream>>>(q_f,  q16,  (int)nBSD);
        cast_f32_to_f16<<<gBSD, blk, 0, stream>>>(k_f,  k16,  (int)nBSD);
        cast_f32_to_f16<<<gBSD, blk, 0, stream>>>(v_f,  v16,  (int)nBSD);
        cast_f32_to_f16<<<gDD,  blk, 0, stream>>>(Wq_f, wq16, (int)nDD);
        cast_f32_to_f16<<<gDD,  blk, 0, stream>>>(Wk_f, wk16, (int)nDD);
        cast_f32_to_f16<<<gDD,  blk, 0, stream>>>(Wv_f, wv16, (int)nDD);
        cast_f32_to_f16<<<gDD,  blk, 0, stream>>>(Wo_f, wo16, (int)nDD);
    }

    // 2) Q/K/V projections (head-split output)
    {
        dim3 grid(MROWS / 64, DMODEL / 128);
        dim3 blk(256);
        gemm_wmma_f16<<<grid, blk, 0, stream>>>(q16, wq16, Qh, nullptr, 0);
        gemm_wmma_f16<<<grid, blk, 0, stream>>>(k16, wk16, Kh, nullptr, 0);
        gemm_wmma_f16<<<grid, blk, 0, stream>>>(v16, wv16, Vh, nullptr, 0);
    }

    // 3) flash attention with ALiBi
    {
        dim3 grid(BATCH * NHEADS, SEQ / 64);
        dim3 blk(128);
        attn_alibi_wmma<<<grid, blk, 0, stream>>>(Qh, Kh, Vh, ctx);
    }

    // 4) output projection (f32 output)
    {
        dim3 grid(MROWS / 64, DMODEL / 128);
        dim3 blk(256);
        gemm_wmma_f16<<<grid, blk, 0, stream>>>(ctx, wo16, nullptr, out, 1);
    }
}
